// BasicTransformer_36318243455248
// MI455X (gfx1250) — matrix operand order repaired and checked by reading, not compiled, not run on hardware
//
#include <hip/hip_runtime.h>

// ---------------- problem constants (from reference) ----------------
#define BB   16
#define CC   128
#define PP   48
#define NN   2304        // PP*PP
#define C8Q  16          // CC/8
#define QKSCALE 0.0883883476483184f   // 1/sqrt(128)

// ---------------- WMMA types ----------------
typedef __attribute__((ext_vector_type(16))) __bf16          v16bf;
typedef __attribute__((ext_vector_type(8)))  float           v8f;
typedef __attribute__((ext_vector_type(8)))  unsigned short  ushort8;
typedef __attribute__((ext_vector_type(4)))  int             v4i;
typedef __attribute__((ext_vector_type(8)))  int             v8i;

union FragBF {
  v16bf   bf;
  ushort8 h[2];     // 2 x 16-byte halves
};

__device__ __forceinline__ unsigned short f2bf(float f) {
  unsigned int u = __float_as_uint(f);
  u = (u + 0x7FFFu + ((u >> 16) & 1u)) >> 16;   // RNE
  return (unsigned short)u;
}

// ---------------- elementwise: f32 -> bf16 (small tensors) ----------------
__global__ void k_f32_to_bf16(const float* __restrict__ src,
                              unsigned short* __restrict__ dst, long long n) {
  long long i = (long long)blockIdx.x * blockDim.x + threadIdx.x;
  if (i < n) dst[i] = f2bf(src[i]);
}

// ---------------- TDM-staged f32 -> bf16 for the big N x N weights --------
// Wave 0 issues a Tensor-Data-Mover 2D tile load (4 rows x NN f32) into LDS,
// waits on TENSORcnt, then all threads convert LDS -> bf16 global.
__global__ void k_w_convert_tdm(const float* __restrict__ src,   // (NN, NN) f32
                                unsigned short* __restrict__ dst) {
  __shared__ float tile[4][NN];
  const int row0 = blockIdx.x * 4;

  if (threadIdx.x < 32) {
    unsigned long long ga = (unsigned long long)(const void*)(src + (size_t)row0 * NN);
    unsigned int lds_off = (unsigned int)(unsigned long long)(void*)&tile[0][0];

    v4i g0;
    g0[0] = 1;                                             // count=1 (valid user D#)
    g0[1] = (int)lds_off;                                  // lds_addr (bytes)
    g0[2] = (int)(ga & 0xffffffffu);                       // global_addr[31:0]
    g0[3] = (int)(((ga >> 32) & 0x01ffffffu) | (2u << 30)); // addr[56:32] | type=2

    v8i g1;
    g1[0] = (int)(2u << 16);                 // workgroup_mask=0, data_size=4B
    g1[1] = (int)((NN & 0xffffu) << 16);     // tensor_dim0[15:0]
    g1[2] = (int)(((NN >> 16) & 0xffffu) | (4u << 16));  // tdim0[31:16] | tensor_dim1[15:0]=4
    g1[3] = (int)((NN & 0xffffu) << 16);     // tensor_dim1[31:16]=0 | tile_dim0=NN
    g1[4] = (int)4;                          // tile_dim1=4, tile_dim2=0
    g1[5] = (int)NN;                         // tensor_dim0_stride[31:0]
    g1[6] = 0;                               // t0s[47:32]=0 | t1s[15:0]=0
    g1[7] = 0;                               // t1s[47:16]=0
    asm volatile("tensor_load_to_lds %0, %1" :: "s"(g0), "s"(g1) : "memory");
    __builtin_amdgcn_s_wait_tensorcnt(0);
  }
  __syncthreads();

  for (int t = threadIdx.x; t < 4 * NN; t += 256) {
    int r = t / NN, c = t % NN;
    dst[(size_t)(row0 + r) * NN + c] = f2bf(tile[r][c]);
  }
}

// ---------------- BatchNorm (training-mode batch stats) ----------------
__global__ void k_bn_stats(const float* __restrict__ x,  // (B, C, N)
                           float* __restrict__ mean, float* __restrict__ rstd) {
  __shared__ float ss[256], sq[256];
  int c = blockIdx.x;
  int tid = threadIdx.x;
  float s = 0.f, q = 0.f;
  const long long total = (long long)BB * NN;
  for (long long t = tid; t < total; t += 256) {
    int bi = (int)(t / NN);
    int n  = (int)(t % NN);
    float v = x[((long long)bi * CC + c) * NN + n];
    s += v; q += v * v;
  }
  ss[tid] = s; sq[tid] = q; __syncthreads();
  for (int o = 128; o > 0; o >>= 1) {
    if (tid < o) { ss[tid] += ss[tid + o]; sq[tid] += sq[tid + o]; }
    __syncthreads();
  }
  if (tid == 0) {
    float m = ss[0] / (float)total;
    float v = sq[0] / (float)total - m * m;
    mean[c] = m;
    rstd[c] = rsqrtf(v + 1e-5f);
  }
}

__global__ void k_bn_apply(const float* __restrict__ x,   // (B, C, N)
                           const float* __restrict__ mean,
                           const float* __restrict__ rstd,
                           const float* __restrict__ g,
                           const float* __restrict__ b,
                           unsigned short* __restrict__ y /* bf16 (B,C,N) */) {
  long long i = (long long)blockIdx.x * blockDim.x + threadIdx.x;
  const long long total = (long long)BB * CC * NN;
  if (i >= total) return;
  int c = (int)((i / NN) % CC);
  float v = (x[i] - mean[c]) * rstd[c] * g[c] + b[c];
  y[i] = f2bf(v);
}

// ---------------- transpose (B,C,N) bf16 -> rows of (B,N,ldk) at coff ----
__global__ void k_transpose_cn_nc(const unsigned short* __restrict__ src, // (B, Cc, Nn)
                                  unsigned short* __restrict__ dst,       // (B, Nn, ldk)
                                  int Cc, int ldk, int coff) {
  long long i = (long long)blockIdx.x * blockDim.x + threadIdx.x;
  long long total = (long long)BB * Cc * NN;
  if (i >= total) return;
  int n = (int)(i % NN);
  int c = (int)((i / NN) % Cc);
  int b = (int)(i / ((long long)Cc * NN));
  dst[((long long)b * NN + n) * ldk + coff + c] = src[i];
}

// ---------------- generic bf16 WMMA GEMM (MT = row-tiles per wave) --------
// Y[b][m][n] = act( sum_k X[m][k] * W[b][n][k] + bias )
// MT=2: wave computes 32x64 (2 A-rows x 4 B-cols = 8 accumulators);
//       block = 8 waves (4m x 2n) = 128 x 128 output tile.
// MT=1: wave computes 16x64 (small-M convs, M=16).
template <int MT>
__global__ void k_gemm_bf16(const unsigned short* __restrict__ X,
                            const unsigned short* __restrict__ W,
                            const float* __restrict__ bias,
                            unsigned short* __restrict__ Ybf,
                            float* __restrict__ Yf,
                            int M, int Nn, int K, int ldw,
                            long long sxb, long long swb, long long syb,
                            int relu, int bias_per_m) {
  const int b    = blockIdx.z;
  const int lane = threadIdx.x & 31;
  const int wave = threadIdx.x >> 5;
  const int wm = wave & 3;
  const int wn = wave >> 2;
  const int m0 = blockIdx.y * 128 + wm * (16 * MT);
  const int n0 = blockIdx.x * 128 + wn * 64;
  if (m0 >= M) return;

  const unsigned short* Xb = X + (long long)b * sxb;
  const unsigned short* Wb = W + (long long)b * swb;

  const int lo  = lane & 15;
  const int lhi = lane >> 4;           // 0 or 1
  const int ak = lhi * 8;              // A half-lane K offset

  v8f acc[MT][4];
#pragma unroll
  for (int i = 0; i < MT; ++i)
#pragma unroll
    for (int j = 0; j < 4; ++j) acc[i][j] = v8f{};

  for (int kb = 0; kb < K; kb += 32) {
    // A fragments: 16x32 (MxK). lane<16: k = kb+0..7, kb+16..23 ; lane>=16: +8
    FragBF a[MT];
    const unsigned short* ap0 = Xb + (long long)(m0 + lo) * K + kb + ak;
    __builtin_prefetch(ap0 + 32, 0, 1);
#pragma unroll
    for (int mtile = 0; mtile < MT; ++mtile) {
      const unsigned short* ap = ap0 + (long long)(16 * mtile) * K;
      a[mtile].h[0] = *(const ushort8*)(ap);
      a[mtile].h[1] = *(const ushort8*)(ap + 16);
    }

#pragma unroll
    for (int sub = 0; sub < 4; ++sub) {
      // B fragment: 32x16 (KxN). lane<16: k = kb+0..7, kb+16..23 ; lane>=16: +8
      FragBF bm;
      const int ncol = n0 + sub * 16 + lo;
      const unsigned short* wp = Wb + (long long)ncol * ldw + kb + lhi * 8;
      bm.h[0] = *(const ushort8*)(wp);
      bm.h[1] = *(const ushort8*)(wp + 16);
#pragma unroll
      for (int mtile = 0; mtile < MT; ++mtile)
        acc[mtile][sub] = __builtin_amdgcn_wmma_f32_16x16x32_bf16(
            false, a[mtile].bf, false, bm.bf, (short)0, acc[mtile][sub], false, false);
    }
  }

  // epilogue: element r of lane: m = m0 + mtile*16 + r + 8*lhi, n = n0+sub*16+lo
#pragma unroll
  for (int mtile = 0; mtile < MT; ++mtile) {
#pragma unroll
    for (int sub = 0; sub < 4; ++sub) {
      const int n = n0 + sub * 16 + lo;
      const float bn = (bias && !bias_per_m) ? bias[n] : 0.f;
#pragma unroll
      for (int r = 0; r < 8; ++r) {
        const int m = m0 + mtile * 16 + r + lhi * 8;
        float y = acc[mtile][sub][r] + bn + ((bias && bias_per_m) ? bias[m] : 0.f);
        if (relu) y = fmaxf(y, 0.f);
        const long long oi = (long long)b * syb + (long long)m * Nn + n;
        if (Ybf) Ybf[oi] = f2bf(y);
        if (Yf)  Yf[oi]  = y;
      }
    }
  }
}

// ---------------- attention pass 1: row max + log-sum-exp ----------------
__global__ void k_attn_stats(const float* __restrict__ qf,   // (B, 16, N)
                             const float* __restrict__ kf,   // (B, 16, N)
                             float* __restrict__ rowm,       // (B, N)
                             float* __restrict__ rowlog) {   // (B, N)
  __shared__ float qs[16][64];
  const int b = blockIdx.y;
  const int tid = threadIdx.x;
  const int i = blockIdx.x * 256 + tid;          // N = 9*256

  float kc[16];
#pragma unroll
  for (int c = 0; c < 16; ++c)
    kc[c] = kf[((long long)b * 16 + c) * NN + i];

  float mx = -1e30f;
  for (int jb = 0; jb < NN; jb += 64) {
    __syncthreads();
    for (int t = tid; t < 1024; t += 256) {
      int c = t >> 6, j = t & 63;
      qs[c][j] = qf[((long long)b * 16 + c) * NN + jb + j];
    }
    __syncthreads();
    for (int j = 0; j < 64; ++j) {
      float s = 0.f;
#pragma unroll
      for (int c = 0; c < 16; ++c) s += kc[c] * qs[c][j];
      mx = fmaxf(mx, s * QKSCALE);
    }
  }
  float l = 0.f;
  for (int jb = 0; jb < NN; jb += 64) {
    __syncthreads();
    for (int t = tid; t < 1024; t += 256) {
      int c = t >> 6, j = t & 63;
      qs[c][j] = qf[((long long)b * 16 + c) * NN + jb + j];
    }
    __syncthreads();
    for (int j = 0; j < 64; ++j) {
      float s = 0.f;
#pragma unroll
      for (int c = 0; c < 16; ++c) s += kc[c] * qs[c][j];
      l += __expf(s * QKSCALE - mx);
    }
  }
  rowm[(long long)b * NN + i] = mx;
  rowlog[(long long)b * NN + i] = __logf(l);
}

// ---------------- attention pass 2: fused P recompute + PV via WMMA -------
__global__ void k_attn_pv(const float* __restrict__ qf,            // (B,16,N)
                          const float* __restrict__ kf,            // (B,16,N)
                          const unsigned short* __restrict__ vbf,  // (B,128,N) bf16
                          const float* __restrict__ rowm,
                          const float* __restrict__ rowlog,
                          unsigned short* __restrict__ catT1,      // (B,N,256)
                          unsigned short* __restrict__ catT2) {    // (B,N,256)
  __shared__ float ks[16][16];            // k columns for this i-tile
  __shared__ float qs[16][32];            // q columns for current j-chunk
  __shared__ unsigned short Ps[16][40];   // P[i][j] tile, bf16, padded rows
  __shared__ float ms[16], ls[16];

  const int b  = blockIdx.y;
  const int i0 = blockIdx.x * 16;
  const int tid  = threadIdx.x;
  const int lane = tid & 31;
  const int wave = tid >> 5;
  const int lo  = lane & 15;
  const int lhi = lane >> 4;
  const int c0 = wave * 16;               // 8 waves x 16 channels = 128

  {
    int c = tid >> 4, ii = tid & 15;
    ks[c][ii] = kf[((long long)b * 16 + c) * NN + i0 + ii];
  }
  if (tid < 16) {
    ms[tid] = rowm[(long long)b * NN + i0 + tid];
    ls[tid] = rowlog[(long long)b * NN + i0 + tid];
  }

  v8f acc = v8f{};

  for (int jb = 0; jb < NN; jb += 32) {
    __syncthreads();                      // Ps/qs from previous iter consumed
    for (int t = 0; t < 2; ++t) {         // stage q chunk: 16 x 32 floats
      int e = tid + t * 256;
      int c = e >> 5, jj = e & 31;
      qs[c][jj] = qf[((long long)b * 16 + c) * NN + jb + jj];
    }
    __syncthreads();
    for (int t = 0; t < 2; ++t) {         // normalized P tile: 2 elems/thread
      int e  = tid + t * 256;
      int ii = e & 15, jj = e >> 4;
      float s = 0.f;
#pragma unroll
      for (int c = 0; c < 16; ++c) s += ks[c][ii] * qs[c][jj];
      float p = __expf(s * QKSCALE - ms[ii] - ls[ii]);
      Ps[ii][jj] = f2bf(p);
    }
    __syncthreads();
    FragBF a;                             // v tile (16c x 32j) from global bf16
    const unsigned short* vp =
        vbf + ((long long)b * CC + c0 + lo) * NN + jb + lhi * 8;
    a.h[0] = *(const ushort8*)(vp);
    a.h[1] = *(const ushort8*)(vp + 16);
    FragBF pm;                            // P (32j x 16i) from LDS
    pm.h[0] = *(const ushort8*)&Ps[lo][lhi * 8];
    pm.h[1] = *(const ushort8*)&Ps[lo][lhi * 8 + 16];
    acc = __builtin_amdgcn_wmma_f32_16x16x32_bf16(
        false, a.bf, false, pm.bf, (short)0, acc, false, false);
  }

  const int i = i0 + lo;                  // store transposed: catT[b][i][128+c]
#pragma unroll
  for (int r = 0; r < 8; ++r) {
    const int c = c0 + r + lhi * 8;
    const unsigned short v = f2bf(acc[r]);
    const long long oi = ((long long)b * NN + i) * 256 + 128 + c;
    catT1[oi] = v;
    catT2[oi] = v;
  }
}

// ==========================================================================
extern "C" void kernel_launch(void* const* d_in, const int* in_sizes, int n_in,
                              void* d_out, int out_size, void* d_ws, size_t ws_size,
                              hipStream_t stream) {
  const float* front_x = (const float*)d_in[0];
  const float* bn1_g   = (const float*)d_in[1];
  const float* bn1_b   = (const float*)d_in[2];
  const float* tm1_w1  = (const float*)d_in[3];
  const float* tm1_b1  = (const float*)d_in[4];
  const float* tm1_w2  = (const float*)d_in[5];
  const float* tm1_b2  = (const float*)d_in[6];
  const float* q_w     = (const float*)d_in[7];
  const float* q_b     = (const float*)d_in[8];
  const float* k_w     = (const float*)d_in[9];
  const float* k_b     = (const float*)d_in[10];
  const float* v_w     = (const float*)d_in[11];
  const float* v_b     = (const float*)d_in[12];
  const float* m1_w    = (const float*)d_in[13];
  const float* m1_b    = (const float*)d_in[14];
  const float* bn2_g   = (const float*)d_in[15];
  const float* bn2_b   = (const float*)d_in[16];
  const float* tm2_w1  = (const float*)d_in[17];
  const float* tm2_b1  = (const float*)d_in[18];
  const float* tm2_w2  = (const float*)d_in[19];
  const float* tm2_b2  = (const float*)d_in[20];
  const float* m2_w    = (const float*)d_in[21];
  const float* m2_b    = (const float*)d_in[22];

  // ---- workspace carve-out ----
  unsigned char* wpc = (unsigned char*)d_ws;
  auto carve = [&](size_t bytes) -> void* {
    void* p = (void*)wpc;
    wpc += (bytes + 255) & ~(size_t)255;
    return p;
  };
  const size_t NB_W = (size_t)NN * NN * 2;        // N x N bf16
  unsigned short* w1bf  = (unsigned short*)carve(NB_W);
  unsigned short* w2bf  = (unsigned short*)carve(NB_W);
  unsigned short* w3bf  = (unsigned short*)carve(NB_W);
  unsigned short* w4bf  = (unsigned short*)carve(NB_W);
  unsigned short* qwbf  = (unsigned short*)carve((size_t)C8Q * CC * 2);
  unsigned short* kwbf  = (unsigned short*)carve((size_t)C8Q * CC * 2);
  unsigned short* vwbf  = (unsigned short*)carve((size_t)CC * CC * 2);
  unsigned short* m1wbf = (unsigned short*)carve((size_t)CC * 2 * CC * 2);
  unsigned short* m2wbf = (unsigned short*)carve((size_t)CC * 2 * CC * 2);
  float* mean1 = (float*)carve(CC * 4);
  float* rstd1 = (float*)carve(CC * 4);
  float* mean2 = (float*)carve(CC * 4);
  float* rstd2 = (float*)carve(CC * 4);
  unsigned short* xhat_bf = (unsigned short*)carve((size_t)BB * CC * NN * 2);
  unsigned short* h_bf    = (unsigned short*)carve((size_t)BB * CC * NN * 2);
  unsigned short* feat_bf = (unsigned short*)carve((size_t)BB * CC * NN * 2);
  unsigned short* v_bf    = (unsigned short*)carve((size_t)BB * CC * NN * 2);
  float* q_f32 = (float*)carve((size_t)BB * C8Q * NN * 4);
  float* k_f32 = (float*)carve((size_t)BB * C8Q * NN * 4);
  float* rowm  = (float*)carve((size_t)BB * NN * 4);
  float* rowlg = (float*)carve((size_t)BB * NN * 4);
  unsigned short* catT1 = (unsigned short*)carve((size_t)BB * NN * 256 * 2);
  unsigned short* catT2 = (unsigned short*)carve((size_t)BB * NN * 256 * 2);
  float* T_f32 = (float*)carve((size_t)BB * CC * NN * 4);

  const dim3 blk(256);
  auto g1 = [&](long long n) { return dim3((unsigned)((n + 255) / 256)); };

  // ---- weight converts: big N x N weights via TDM-staged LDS path ----
  k_w_convert_tdm<<<dim3(NN / 4), blk, 0, stream>>>(tm1_w1, w1bf);
  k_w_convert_tdm<<<dim3(NN / 4), blk, 0, stream>>>(tm1_w2, w2bf);
  k_w_convert_tdm<<<dim3(NN / 4), blk, 0, stream>>>(tm2_w1, w3bf);
  k_w_convert_tdm<<<dim3(NN / 4), blk, 0, stream>>>(tm2_w2, w4bf);
  k_f32_to_bf16<<<g1(C8Q * CC), blk, 0, stream>>>(q_w, qwbf, C8Q * CC);
  k_f32_to_bf16<<<g1(C8Q * CC), blk, 0, stream>>>(k_w, kwbf, C8Q * CC);
  k_f32_to_bf16<<<g1(CC * CC), blk, 0, stream>>>(v_w, vwbf, CC * CC);
  k_f32_to_bf16<<<g1(CC * 2 * CC), blk, 0, stream>>>(m1_w, m1wbf, CC * 2 * CC);
  k_f32_to_bf16<<<g1(CC * 2 * CC), blk, 0, stream>>>(m2_w, m2wbf, CC * 2 * CC);

  const long long ACT = (long long)BB * CC * NN;

  // ---- BN1 + TransformModule 1 ----
  k_bn_stats<<<dim3(CC), blk, 0, stream>>>(front_x, mean1, rstd1);
  k_bn_apply<<<g1(ACT), blk, 0, stream>>>(front_x, mean1, rstd1, bn1_g, bn1_b, xhat_bf);

  dim3 gTM(NN / 128, (BB * CC) / 128, 1);   // 18 x 16
  k_gemm_bf16<2><<<gTM, blk, 0, stream>>>(xhat_bf, w1bf, tm1_b1, h_bf, nullptr,
                                          BB * CC, NN, NN, NN, 0, 0, 0, 1, 0);
  k_gemm_bf16<2><<<gTM, blk, 0, stream>>>(h_bf, w2bf, tm1_b2, feat_bf, nullptr,
                                          BB * CC, NN, NN, NN, 0, 0, 0, 1, 0);

  // ---- features^T into catT1[:, 0:128] ----
  k_transpose_cn_nc<<<g1(ACT), blk, 0, stream>>>(feat_bf, catT1, CC, 256, 0);

  // ---- q / k / v 1x1 convs (batched GEMMs over featT rows of catT1) ----
  dim3 gQK(NN / 128, 1, BB);
  k_gemm_bf16<1><<<gQK, blk, 0, stream>>>(qwbf, catT1, q_b, nullptr, q_f32,
                                          C8Q, NN, CC, 256, 0, (long long)NN * 256,
                                          (long long)C8Q * NN, 0, 1);
  k_gemm_bf16<1><<<gQK, blk, 0, stream>>>(kwbf, catT1, k_b, nullptr, k_f32,
                                          C8Q, NN, CC, 256, 0, (long long)NN * 256,
                                          (long long)C8Q * NN, 0, 1);
  k_gemm_bf16<2><<<gQK, blk, 0, stream>>>(vwbf, catT1, v_b, v_bf, nullptr,
                                          CC, NN, CC, 256, 0, (long long)NN * 256,
                                          (long long)CC * NN, 0, 1);

  // ---- attention: softmax stats, then fused PV (writes catT1/2[:,128:256]) ----
  k_attn_stats<<<dim3(NN / 256, BB), blk, 0, stream>>>(q_f32, k_f32, rowm, rowlg);
  k_attn_pv<<<dim3(NN / 16, BB), blk, 0, stream>>>(q_f32, k_f32, v_bf, rowm, rowlg,
                                                   catT1, catT2);

  // ---- m1 conv over concat -> T (f32) ----
  dim3 gM(NN / 128, 1, BB);
  k_gemm_bf16<2><<<gM, blk, 0, stream>>>(m1wbf, catT1, m1_b, nullptr, T_f32,
                                         CC, NN, 2 * CC, 256, 0, (long long)NN * 256,
                                         (long long)CC * NN, 0, 1);

  // ---- BN2 + TransformModule 2 ----
  k_bn_stats<<<dim3(CC), blk, 0, stream>>>(T_f32, mean2, rstd2);
  k_bn_apply<<<g1(ACT), blk, 0, stream>>>(T_f32, mean2, rstd2, bn2_g, bn2_b, xhat_bf);
  k_gemm_bf16<2><<<gTM, blk, 0, stream>>>(xhat_bf, w3bf, tm2_b1, h_bf, nullptr,
                                          BB * CC, NN, NN, NN, 0, 0, 0, 1, 0);
  k_gemm_bf16<2><<<gTM, blk, 0, stream>>>(h_bf, w4bf, tm2_b2, feat_bf, nullptr,
                                          BB * CC, NN, NN, NN, 0, 0, 0, 1, 0);

  // ---- front_res^T into catT2[:, 0:128]; final m2 conv -> d_out (f32) ----
  k_transpose_cn_nc<<<g1(ACT), blk, 0, stream>>>(feat_bf, catT2, CC, 256, 0);
  k_gemm_bf16<2><<<gM, blk, 0, stream>>>(m2wbf, catT2, m2_b, nullptr, (float*)d_out,
                                         CC, NN, 2 * CC, 256, 0, (long long)NN * 256,
                                         (long long)CC * NN, 0, 1);
}
